// Attention_51539608123
// MI455X (gfx1250) — matrix operand order repaired and checked by reading, not compiled, not run on hardware
//
#include <hip/hip_runtime.h>
#include <hip/hip_bf16.h>
#include <math.h>

// ---------------------------------------------------------------------------
// Types for CDNA5 WMMA (wave32): V_WMMA_F32_16X16X32_BF16
// ---------------------------------------------------------------------------
typedef __bf16 bf16_t;
typedef __attribute__((ext_vector_type(8)))  __bf16 v8bf;
typedef __attribute__((ext_vector_type(16))) __bf16 v16bf;
typedef __attribute__((ext_vector_type(8)))  float  v8f;

static __device__ __forceinline__ v8f wmma_bf16(v16bf a, v16bf b, v8f c) {
  // (neg_a, A, neg_b, B, c_mod, C, reuse_a, reuse_b)
  return __builtin_amdgcn_wmma_f32_16x16x32_bf16(false, a, false, b, (short)0, c,
                                                 false, false);
}

static __device__ __forceinline__ v16bf cat8(v8bf lo, v8bf hi) {
  return __builtin_shufflevector(lo, hi, 0, 1, 2, 3, 4, 5, 6, 7, 8, 9, 10, 11,
                                 12, 13, 14, 15);
}

// ---------------------------------------------------------------------------
// CDNA5 async global->LDS copy (ASYNCcnt-tracked). The low 32 bits of a
// generic pointer to __shared__ are the wave-relative LDS byte offset
// (ISA 10.2).
// ---------------------------------------------------------------------------
static __device__ __forceinline__ void async_copy_b128(const bf16_t* g,
                                                       bf16_t* l) {
  unsigned lds_off = (unsigned)(uintptr_t)l;
  asm volatile("global_load_async_to_lds_b128 %0, %1, off"
               :
               : "v"(lds_off), "v"(g)
               : "memory");
}
static __device__ __forceinline__ void async_wait0() {
  asm volatile("s_wait_asynccnt 0x0" ::: "memory");
}

// ---------------------------------------------------------------------------
// f32 -> bf16 convert
// ---------------------------------------------------------------------------
__global__ __launch_bounds__(256) void cvt_f32_to_bf16(
    const float* __restrict__ src, bf16_t* __restrict__ dst, int n) {
  int i = blockIdx.x * 256 + threadIdx.x;
  if (i < n) dst[i] = (bf16_t)src[i];
}

// ---------------------------------------------------------------------------
// Double-buffered WMMA GEMM, compile-time dims:
//   C[M,N](f32) = A[M,K](bf16 rm) * B[K,N](bf16 rm)
// Block tile 128x128, K-step 32 (== WMMA K). 256 threads = 8 waves,
// wave grid 4(M) x 2(N); each wave computes 32x64 = 2x4 WMMA 16x16 tiles.
// Tile t+1 is staged (A via global_load_async_to_lds_b128, B via VGPRs with
// a transpose scatter placed after the WMMA stage) while tile t computes.
// ---------------------------------------------------------------------------
#define GBM 128
#define GBN 128
#define GBK 32

template <int M, int N, int K>
__global__ __launch_bounds__(256) void gemm_bf16_rrr(
    const bf16_t* __restrict__ A, const bf16_t* __restrict__ Bm,
    float* __restrict__ C) {
  __shared__ __align__(16) bf16_t As[2][GBM][GBK];      // [m][k]
  __shared__ __align__(16) bf16_t Bs[2][GBN][GBK + 8];  // [n][k] transposed

  const int tid = threadIdx.x;
  const int lane = tid & 31;
  const int wave = tid >> 5;
  const int wm = wave >> 1;  // 0..3
  const int wn = wave & 1;   // 0..1
  const int lm = lane & 15;
  const int lh = lane >> 4;
  const int bm = blockIdx.y * GBM;
  const int bn = blockIdx.x * GBN;

  const int ar = tid >> 1, aks = (tid & 1) << 4;
  const int bkr = tid >> 3, bnb = (tid & 7) << 4;
  constexpr int NT = K / GBK;

  // Prologue: stage tile 0
  {
    const bf16_t* asrc = A + (size_t)(bm + ar) * K + aks;
    async_copy_b128(asrc, &As[0][ar][aks]);
    async_copy_b128(asrc + 8, &As[0][ar][aks + 8]);
    const bf16_t* bsrc = Bm + (size_t)bkr * N + bn + bnb;
    v8bf b0 = *(const v8bf*)bsrc;
    v8bf b1 = *(const v8bf*)(bsrc + 8);
#pragma unroll
    for (int i = 0; i < 8; ++i) {
      Bs[0][bnb + i][bkr] = b0[i];
      Bs[0][bnb + 8 + i][bkr] = b1[i];
    }
  }
  async_wait0();
  __syncthreads();

  v8f acc[2][4] = {};

  for (int t = 0; t < NT; ++t) {
    const int cur = t & 1, nxt = cur ^ 1;
    const bool more = (t + 1) < NT;
    v8bf nb0, nb1;
    if (more) {
      const int k0 = (t + 1) * GBK;
      const bf16_t* asrc = A + (size_t)(bm + ar) * K + k0 + aks;
      async_copy_b128(asrc, &As[nxt][ar][aks]);
      async_copy_b128(asrc + 8, &As[nxt][ar][aks + 8]);
      const bf16_t* bsrc = Bm + (size_t)(k0 + bkr) * N + bn + bnb;
      nb0 = *(const v8bf*)bsrc;
      nb1 = *(const v8bf*)(bsrc + 8);
    }

    // A fragment: lane lm = row M; lh=0 -> K {0..7,16..23}, lh=1 -> {8..15,24..31}
    v16bf afr[2];
#pragma unroll
    for (int mi = 0; mi < 2; ++mi) {
      const bf16_t* p0 = &As[cur][wm * 32 + mi * 16 + lm][lh * 8];
      afr[mi] = cat8(*(const v8bf*)p0, *(const v8bf*)(p0 + 16));
    }
    // B fragment: lane lm = column N; lh=0 -> K {0..7,16..23}, lh=1 -> {8..15,24..31}
    v16bf bfr[4];
#pragma unroll
    for (int ni = 0; ni < 4; ++ni) {
      const bf16_t* p0 = &Bs[cur][wn * 64 + ni * 16 + lm][lh * 8];
      bfr[ni] = cat8(*(const v8bf*)p0, *(const v8bf*)(p0 + 16));
    }
#pragma unroll
    for (int mi = 0; mi < 2; ++mi)
#pragma unroll
      for (int ni = 0; ni < 4; ++ni)
        acc[mi][ni] = wmma_bf16(afr[mi], bfr[ni], acc[mi][ni]);

    // Transpose-scatter next B tile after the WMMA stage (latency hidden)
    if (more) {
#pragma unroll
      for (int i = 0; i < 8; ++i) {
        Bs[nxt][bnb + i][bkr] = nb0[i];
        Bs[nxt][bnb + 8 + i][bkr] = nb1[i];
      }
    }
    async_wait0();
    __syncthreads();
  }

  // C/D layout: VGPR r -> M = r + 8*lh, N = lm.
  // One base pointer per wave; all 64 offsets are compile-time constants.
  float* cbase = C + (size_t)(bm + wm * 32 + 8 * lh) * N + bn + wn * 64 + lm;
#pragma unroll
  for (int mi = 0; mi < 2; ++mi)
#pragma unroll
    for (int ni = 0; ni < 4; ++ni)
#pragma unroll
      for (int r = 0; r < 8; ++r)
        cbase[(mi * 16 + r) * N + ni * 16] = acc[mi][ni][r];
}

// ---------------------------------------------------------------------------
// RoPE for Q: qf [B,S,H,D] f32 -> qb [B,H,S,D] bf16
// ---------------------------------------------------------------------------
__global__ __launch_bounds__(256) void rope_q_kernel(
    const float* __restrict__ qf, const float* __restrict__ cs,
    const float* __restrict__ sn, bf16_t* __restrict__ qb, int B, int S, int H,
    int D) {
  int idx = blockIdx.x * 256 + threadIdx.x;
  int half = D >> 1;
  int total = B * S * H * half;
  if (idx >= total) return;
  int dh = idx % half;
  int t = idx / half;
  int hh = t % H;
  t /= H;
  int s = t % S;
  int bb = t / S;
  const float* src = qf + ((size_t)(bb * S + s) * H + hh) * D;
  float c = cs[s * half + dh], si = sn[s * half + dh];
  float a = src[dh], b = src[dh + half];
  float r1 = a * c - b * si;
  float r2 = b * c + a * si;
  bf16_t* dst = qb + ((size_t)(bb * H + hh) * S + s) * D;
  dst[dh] = (bf16_t)r1;
  dst[dh + half] = (bf16_t)r2;
}

// ---------------------------------------------------------------------------
// RoPE for K: kf [B,S,Hkv,D] f32 -> kb [B,Hkv,S,D] bf16
//             + GQA-expanded f32 kout [B,Hkv*G,S,D] (returned output)
// ---------------------------------------------------------------------------
__global__ __launch_bounds__(256) void rope_k_kernel(
    const float* __restrict__ kf, const float* __restrict__ cs,
    const float* __restrict__ sn, bf16_t* __restrict__ kb,
    float* __restrict__ kout, int B, int S, int Hkv, int G, int D) {
  int idx = blockIdx.x * 256 + threadIdx.x;
  int half = D >> 1;
  int total = B * S * Hkv * half;
  if (idx >= total) return;
  int dh = idx % half;
  int t = idx / half;
  int hh = t % Hkv;
  t /= Hkv;
  int s = t % S;
  int bb = t / S;
  const float* src = kf + ((size_t)(bb * S + s) * Hkv + hh) * D;
  float c = cs[s * half + dh], si = sn[s * half + dh];
  float a = src[dh], b = src[dh + half];
  float r1 = a * c - b * si;
  float r2 = b * c + a * si;
  bf16_t* dst = kb + ((size_t)(bb * Hkv + hh) * S + s) * D;
  dst[dh] = (bf16_t)r1;
  dst[dh + half] = (bf16_t)r2;
  int Hq = Hkv * G;
  for (int g = 0; g < G; ++g) {
    float* d2 = kout + ((size_t)(bb * Hq + hh * G + g) * S + s) * D;
    d2[dh] = r1;
    d2[dh + half] = r2;
  }
}

// ---------------------------------------------------------------------------
// V pack: vf [B,S,Hkv,D] f32 -> vb [B,Hkv,S,D] bf16 + expanded f32 vout
// ---------------------------------------------------------------------------
__global__ __launch_bounds__(256) void v_pack_kernel(
    const float* __restrict__ vf, bf16_t* __restrict__ vb,
    float* __restrict__ vout, int B, int S, int Hkv, int G, int D) {
  int idx = blockIdx.x * 256 + threadIdx.x;
  int total = B * S * Hkv * D;
  if (idx >= total) return;
  int d = idx % D;
  int t = idx / D;
  int hh = t % Hkv;
  t /= Hkv;
  int s = t % S;
  int bb = t / S;
  float val = vf[((size_t)(bb * S + s) * Hkv + hh) * D + d];
  vb[((size_t)(bb * Hkv + hh) * S + s) * D + d] = (bf16_t)val;
  int Hq = Hkv * G;
  for (int g = 0; g < G; ++g)
    vout[((size_t)(bb * Hq + hh * G + g) * S + s) * D + d] = val;
}

// ---------------------------------------------------------------------------
// Flash attention forward (no 1/sqrt(d) scaling, additive mask, bf16 WMMA).
// Block: 128 query rows, 8 waves; each wave owns 16 query rows.
// Q fragments are loaded ONCE from global into registers (A operand layout is
// row-per-lane) and reused for all key blocks. K/V tiles are double-buffered:
// K via async global->LDS DMA, V via VGPRs with transpose scatter placed
// after the WMMA stage. Online softmax in f32.
// Q [B,Hq,S,D] bf16, K/V [B,Hkv,S,D] bf16, O [B,S,Hq,D] bf16.
// ---------------------------------------------------------------------------
__global__ __launch_bounds__(256) void attn_fwd(
    const bf16_t* __restrict__ Q, const bf16_t* __restrict__ Kc,
    const bf16_t* __restrict__ Vc, const float* __restrict__ mask,
    bf16_t* __restrict__ O) {
  constexpr int S = 2048, Hq = 16, Hkv = 4, D = 128;
  constexpr int AQ = 128, AKB = 64, NT = S / AKB;

  __shared__ __align__(16) bf16_t Ks[2][AKB][D];   // 2 x 16 KB [key][d]
  __shared__ __align__(16) bf16_t Vt[2][D][AKB];   // 2 x 16 KB [d][key]
  __shared__ __align__(16) bf16_t Ps[AQ][AKB];     // 16 KB     [q][key]

  const int tid = threadIdx.x;
  const int lane = tid & 31;
  const int wave = tid >> 5;
  const int lm = lane & 15;
  const int lh = lane >> 4;
  const int qb0 = blockIdx.x * AQ;
  const int h = blockIdx.y;
  const int b = blockIdx.z;
  const int hk = h >> 2;  // GQA_GROUPS = 4

  const bf16_t* Qp = Q + ((size_t)(b * Hq + h) * S) * D;
  const bf16_t* Kp = Kc + ((size_t)(b * Hkv + hk) * S) * D;
  const bf16_t* Vp = Vc + ((size_t)(b * Hkv + hk) * S) * D;

  // Q fragments: loaded once, straight from global (row-per-lane layout).
  v16bf qfr[4];
  {
    const bf16_t* qrow = Qp + (size_t)(qb0 + wave * 16 + lm) * D;
#pragma unroll
    for (int kd = 0; kd < 4; ++kd) {
      const bf16_t* p0 = qrow + kd * 32 + lh * 8;
      qfr[kd] = cat8(*(const v8bf*)p0, *(const v8bf*)(p0 + 16));
    }
  }

  const int lr = tid >> 2, lcs = (tid & 3) * 32;

  // Prologue: stage key-block 0
  {
    const bf16_t* ksrc = Kp + (size_t)lr * D + lcs;
#pragma unroll
    for (int i = 0; i < 4; ++i)
      async_copy_b128(ksrc + i * 8, &Ks[0][lr][lcs + i * 8]);
    const bf16_t* vsrc = Vp + (size_t)lr * D + lcs;
#pragma unroll
    for (int i = 0; i < 4; ++i) {
      v8bf v = *(const v8bf*)(vsrc + i * 8);
#pragma unroll
      for (int j = 0; j < 8; ++j) Vt[0][lcs + i * 8 + j][lr] = v[j];
    }
  }
  async_wait0();
  __syncthreads();

  v8f oacc[8] = {};
  float mrow[8], lrow[8];
#pragma unroll
  for (int r = 0; r < 8; ++r) {
    mrow[r] = -1e30f;
    lrow[r] = 0.f;
  }

  for (int t = 0; t < NT; ++t) {
    const int cur = t & 1, nxt = cur ^ 1;
    const int kb = t * AKB;
    const bool more = (t + 1) < NT;
    v8bf vregs[4];
    if (more) {
      const bf16_t* ksrc = Kp + (size_t)(kb + AKB + lr) * D + lcs;
#pragma unroll
      for (int i = 0; i < 4; ++i)
        async_copy_b128(ksrc + i * 8, &Ks[nxt][lr][lcs + i * 8]);
      const bf16_t* vsrc = Vp + (size_t)(kb + AKB + lr) * D + lcs;
#pragma unroll
      for (int i = 0; i < 4; ++i) vregs[i] = *(const v8bf*)(vsrc + i * 8);
    }

    // Scores S = Q * K^T (per wave: 16 rows x 64 keys)
    v8f sacc[4] = {};
#pragma unroll
    for (int ni = 0; ni < 4; ++ni)
#pragma unroll
      for (int kd = 0; kd < 4; ++kd) {
        const bf16_t* bp = &Ks[cur][ni * 16 + lm][kd * 32 + lh * 8];
        v16bf bfr = cat8(*(const v8bf*)bp, *(const v8bf*)(bp + 16));
        sacc[ni] = wmma_bf16(qfr[kd], bfr, sacc[ni]);
      }

    // Additive mask (faithful to reference; zeros live in L2)
    {
      const float* mbase =
          mask + (size_t)(qb0 + wave * 16 + 8 * lh) * S + kb + lm;
#pragma unroll
      for (int ni = 0; ni < 4; ++ni)
#pragma unroll
        for (int r = 0; r < 8; ++r) sacc[ni][r] += mbase[r * S + ni * 16];
    }

    // Online softmax (row N-dim is spread across a 16-lane half)
#pragma unroll
    for (int r = 0; r < 8; ++r) {
      float mx = sacc[0][r];
#pragma unroll
      for (int ni = 1; ni < 4; ++ni) mx = fmaxf(mx, sacc[ni][r]);
#pragma unroll
      for (int off = 1; off < 16; off <<= 1)
        mx = fmaxf(mx, __shfl_xor(mx, off, 32));
      float mnew = fmaxf(mrow[r], mx);
      float corr = __expf(mrow[r] - mnew);
      float sum = 0.f;
#pragma unroll
      for (int ni = 0; ni < 4; ++ni) {
        float pv = __expf(sacc[ni][r] - mnew);
        sacc[ni][r] = pv;
        sum += pv;
      }
#pragma unroll
      for (int off = 1; off < 16; off <<= 1) sum += __shfl_xor(sum, off, 32);
      lrow[r] = lrow[r] * corr + sum;
      mrow[r] = mnew;
#pragma unroll
      for (int ni = 0; ni < 8; ++ni) oacc[ni][r] *= corr;
    }

    // Stage P to LDS (same-wave region only: per-wave LDS ordering suffices)
#pragma unroll
    for (int ni = 0; ni < 4; ++ni)
#pragma unroll
      for (int r = 0; r < 8; ++r)
        Ps[wave * 16 + r + 8 * lh][ni * 16 + lm] = (bf16_t)sacc[ni][r];

    // O += P * V
#pragma unroll
    for (int kc = 0; kc < 2; ++kc) {  // key chunks of 32
      const bf16_t* ap = &Ps[wave * 16 + lm][kc * 32 + lh * 8];
      v16bf afr = cat8(*(const v8bf*)ap, *(const v8bf*)(ap + 16));
#pragma unroll
      for (int ni = 0; ni < 8; ++ni) {
        const bf16_t* bp = &Vt[cur][ni * 16 + lm][kc * 32 + lh * 8];
        v16bf bfr = cat8(*(const v8bf*)bp, *(const v8bf*)(bp + 16));
        oacc[ni] = wmma_bf16(afr, bfr, oacc[ni]);
      }
    }

    // Transpose-scatter next V tile after the WMMA stage (latency hidden)
    if (more) {
#pragma unroll
      for (int i = 0; i < 4; ++i)
#pragma unroll
        for (int j = 0; j < 8; ++j) Vt[nxt][lcs + i * 8 + j][lr] = vregs[i][j];
    }
    async_wait0();
    __syncthreads();
  }

  // Epilogue: normalize, write O as [B,S,Hq,D] bf16 (feeds output GEMM).
  {
    bf16_t* obase =
        O + ((size_t)b * S + qb0 + wave * 16 + 8 * lh) * (Hq * D) + h * D + lm;
#pragma unroll
    for (int ni = 0; ni < 8; ++ni)
#pragma unroll
      for (int r = 0; r < 8; ++r) {
        float val = oacc[ni][r] / lrow[r];
        obase[r * (Hq * D) + ni * 16] = (bf16_t)val;
      }
  }
}

// ---------------------------------------------------------------------------
// Host-side launch
// ---------------------------------------------------------------------------
extern "C" void kernel_launch(void* const* d_in, const int* in_sizes, int n_in,
                              void* d_out, int out_size, void* d_ws,
                              size_t ws_size, hipStream_t stream) {
  (void)in_sizes;
  (void)n_in;
  (void)out_size;
  (void)ws_size;
  const int B = 2, S = 2048, E = 2048, Hq = 16, Hkv = 4, D = 128, G = 4;

  const float* x = (const float*)d_in[0];
  const float* rc = (const float*)d_in[1];
  const float* rs = (const float*)d_in[2];
  const float* mask = (const float*)d_in[3];
  const float* Wq = (const float*)d_in[4];
  const float* Wk = (const float*)d_in[5];
  const float* Wv = (const float*)d_in[6];
  const float* Wo = (const float*)d_in[7];

  float* out = (float*)d_out;
  float* kout = out + (size_t)B * S * E;
  float* vout = kout + (size_t)B * Hq * S * D;

  char* p = (char*)d_ws;
  auto carve = [&](size_t bytes) -> char* {
    char* q = p;
    p += (bytes + 255) & ~(size_t)255;
    return q;
  };
  const size_t BS = (size_t)B * S;  // 4096
  bf16_t* xb = (bf16_t*)carve(BS * E * 2);
  bf16_t* wqb = (bf16_t*)carve((size_t)E * Hq * D * 2);
  bf16_t* wkb = (bf16_t*)carve((size_t)E * Hkv * D * 2);
  bf16_t* wvb = (bf16_t*)carve((size_t)E * Hkv * D * 2);
  bf16_t* wob = (bf16_t*)carve((size_t)Hq * D * E * 2);
  float* qf = (float*)carve(BS * Hq * D * 4);
  float* kf = (float*)carve(BS * Hkv * D * 4);
  float* vf = (float*)carve(BS * Hkv * D * 4);
  bf16_t* qb = (bf16_t*)carve(BS * Hq * D * 2);
  bf16_t* kb = (bf16_t*)carve(BS * Hkv * D * 2);
  bf16_t* vb = (bf16_t*)carve(BS * Hkv * D * 2);
  bf16_t* ob = (bf16_t*)carve(BS * Hq * D * 2);

  auto cvt = [&](const float* s, bf16_t* d, int n) {
    cvt_f32_to_bf16<<<(n + 255) / 256, 256, 0, stream>>>(s, d, n);
  };
  cvt(x, xb, (int)(BS * E));
  cvt(Wq, wqb, E * Hq * D);
  cvt(Wk, wkb, E * Hkv * D);
  cvt(Wv, wvb, E * Hkv * D);
  cvt(Wo, wob, Hq * D * E);

  // QKV projections (compile-time dims)
  {
    dim3 g(2048 / GBN, 4096 / GBM);
    gemm_bf16_rrr<4096, 2048, 2048><<<g, 256, 0, stream>>>(xb, wqb, qf);
  }
  {
    dim3 g(512 / GBN, 4096 / GBM);
    gemm_bf16_rrr<4096, 512, 2048><<<g, 256, 0, stream>>>(xb, wkb, kf);
  }
  {
    dim3 g(512 / GBN, 4096 / GBM);
    gemm_bf16_rrr<4096, 512, 2048><<<g, 256, 0, stream>>>(xb, wvb, vf);
  }

  // RoPE + layout + returned k/v outputs
  rope_q_kernel<<<(B * S * Hq * (D / 2) + 255) / 256, 256, 0, stream>>>(
      qf, rc, rs, qb, B, S, Hq, D);
  rope_k_kernel<<<(B * S * Hkv * (D / 2) + 255) / 256, 256, 0, stream>>>(
      kf, rc, rs, kb, kout, B, S, Hkv, G, D);
  v_pack_kernel<<<(B * S * Hkv * D + 255) / 256, 256, 0, stream>>>(
      vf, vb, vout, B, S, Hkv, G, D);

  // Attention
  {
    dim3 g(S / 128, Hq, B);
    attn_fwd<<<g, 256, 0, stream>>>(qb, kb, vb, mask, ob);
  }

  // Output projection -> out
  {
    dim3 g(2048 / GBN, 4096 / GBM);
    gemm_bf16_rrr<4096, 2048, 2048><<<g, 256, 0, stream>>>(ob, wob, out);
  }
}
